// ProjEnc_28956669510151
// MI455X (gfx1250) — matrix operand order repaired and checked by reading, not compiled, not run on hardware
//
#include <hip/hip_runtime.h>
#include <hip/hip_bf16.h>
#include <math.h>
#include <stdint.h>

// ---- problem constants (match reference) ----
constexpr int B_ = 4;
constexpr int N_ = 4096;
constexpr int K_ = 32;     // neighbors == conv "W" dimension
constexpr int C_ = 64;     // GRAPH channels
constexpr float EPS_ = 1e-5f;

typedef _Float16 v16h __attribute__((ext_vector_type(16)));
typedef _Float16 h8   __attribute__((ext_vector_type(8)));
typedef float    v8f  __attribute__((ext_vector_type(8)));

// Build an A-fragment (16x32 f16, M = lane&15, K-halves at hi*8 and 16+hi*8):
// reads elements [base .. base+7] and [base+16 .. base+23] (contiguous f16).
__device__ __forceinline__ v16h load_afrag(const _Float16* base) {
    h8 a0 = *(const h8*)(base);
    h8 a1 = *(const h8*)(base + 16);
    return __builtin_shufflevector(a0, a1, 0,1,2,3,4,5,6,7,8,9,10,11,12,13,14,15);
}

// B-fragment: 16 contiguous f16 (K-halves), only 16B alignment needed.
__device__ __forceinline__ v16h load_bfrag(const _Float16* base) {
    h8 b0 = *(const h8*)(base);
    h8 b1 = *(const h8*)(base + 8);
    return __builtin_shufflevector(b0, b1, 0,1,2,3,4,5,6,7,8,9,10,11,12,13,14,15);
}

__device__ __forceinline__ v8f wmma16(v16h a, v16h b, v8f c) {
    return __builtin_amdgcn_wmma_f32_16x16x32_f16(false, a, false, b, (short)0, c,
                                                  false, false);
}

// ---------------------------------------------------------------------------
// TDM: contiguous f16 copy global -> LDS via tensor_load_to_lds (TENSORcnt).
// 1-D tile descriptor per CDNA5 ISA:
//   g0 = { count=1, lds_addr, global_addr[31:0], global_addr[56:32] | type=2 }
//   g1 = { data_size=2B, tensor_dim0=cnt, tensor_dim1=1, tile_dim0=cnt,
//          tile_dim1=1, strides=cnt }
// cnt must be < 65536 f16 elements.
// ---------------------------------------------------------------------------
__device__ __forceinline__ void tdm_copy_f16(const void* gsrc, void* lds_dst,
                                             unsigned cnt)
{
    typedef unsigned int u32x4 __attribute__((ext_vector_type(4)));
    typedef int i32x4 __attribute__((ext_vector_type(4)));
    typedef int i32x8 __attribute__((ext_vector_type(8)));
    unsigned long long ga = (unsigned long long)(uintptr_t)gsrc;
    unsigned lds_addr = (unsigned)(uintptr_t)lds_dst;   // low 32b = LDS byte addr
    u32x4 g0;
    g0[0] = 1u;                                          // count=1, user mode
    g0[1] = lds_addr;
    g0[2] = (unsigned)(ga & 0xFFFFFFFFu);
    g0[3] = (unsigned)((ga >> 32) & 0x01FFFFFFu) | 0x80000000u;  // type=2
    i32x8 g1;
    g1[0] = (int)(1u << 16);                             // data_size = 2 bytes
    g1[1] = (int)((cnt & 0xFFFFu) << 16);                // tensor_dim0 lo16
    g1[2] = (int)(((cnt >> 16) & 0xFFFFu) | (1u << 16)); // td0 hi | tensor_dim1=1
    g1[3] = (int)((cnt & 0xFFFFu) << 16);                // tile_dim0 = cnt
    g1[4] = 1;                                           // tile_dim1=1, tile_dim2=0
    g1[5] = (int)cnt;                                    // tensor_dim0_stride lo32
    g1[6] = (int)((cnt & 0xFFFFu) << 16);                // str0 hi=0 | str1 lo16
    g1[7] = (int)((cnt >> 16) & 0xFFFFu);                // str1 hi
    i32x4 z4; z4[0] = z4[1] = z4[2] = z4[3] = 0;
#if defined(__clang_major__) && (__clang_major__ >= 23)
    i32x8 z8;
    for (int i = 0; i < 8; ++i) z8[i] = 0;
    __builtin_amdgcn_tensor_load_to_lds(g0, g1, z4, z4, z8, 0);
#else
    __builtin_amdgcn_tensor_load_to_lds(g0, g1, z4, z4, 0);
#endif
}

// ---------------------------------------------------------------------------
// K0a: convert conv weights HWIO (tap,ci,co) -> f16 [tap][co][ci]; w_blk -> f16
// ---------------------------------------------------------------------------
__global__ __launch_bounds__(256) void prep_weights_kernel(
    const float* __restrict__ wc1, const float* __restrict__ wc2,
    const float* __restrict__ wblk,
    _Float16* __restrict__ wc1h, _Float16* __restrict__ wc2h,
    _Float16* __restrict__ wblkh)
{
    int e = blockIdx.x * 256 + threadIdx.x;
    if (e < 36864) {                       // conv1: dest [tap][co][ci]
        int tap = e / 4096, r = e % 4096, co = r / 64, ci = r % 64;
        wc1h[e] = (_Float16)wc1[(tap * 64 + ci) * 64 + co];
    } else if (e < 73728) {
        int e2 = e - 36864;
        int tap = e2 / 4096, r = e2 % 4096, co = r / 64, ci = r % 64;
        wc2h[e2] = (_Float16)wc2[(tap * 64 + ci) * 64 + co];
    } else if (e < 77824) {                // w_blk already [d][c]
        int e2 = e - 73728;
        wblkh[e2] = (_Float16)wblk[e2];
    }
}

// ---------------------------------------------------------------------------
// K0b: fold BN into scale/shift; zero GN stats accumulators
// ---------------------------------------------------------------------------
__global__ void prep_bn_kernel(
    const float* g1, const float* b1, const float* m1, const float* v1,
    const float* g2, const float* b2, const float* m2, const float* v2,
    float* bnp, float* stats)
{
    int t = threadIdx.x;
    if (t < 64) {
        float s1 = g1[t] * rsqrtf(v1[t] + EPS_);
        bnp[t]        = s1;
        bnp[64 + t]   = b1[t] - m1[t] * s1;
        float s2 = g2[t] * rsqrtf(v2[t] + EPS_);
        bnp[128 + t]  = s2;
        bnp[192 + t]  = b2[t] - m2[t] * s2;
    }
    if (t >= 64 && t < 96) stats[t - 64] = 0.f;  // 32 floats of GN stats
}

// ---------------------------------------------------------------------------
// K1: f = pc . W_in^T + b_in  (B,N,8)  and squared norms
// ---------------------------------------------------------------------------
__global__ __launch_bounds__(256) void fsq_kernel(
    const float* __restrict__ opc, const float* __restrict__ w_in,
    const float* __restrict__ b_in, float* __restrict__ f,
    float* __restrict__ sqn)
{
    int i = blockIdx.x * 256 + threadIdx.x;     // b*N + n
    if (i >= B_ * N_) return;
    float x = opc[i * 3 + 0], y = opc[i * 3 + 1], z = opc[i * 3 + 2];
    sqn[i] = x * x + y * y + z * z;
#pragma unroll
    for (int d = 0; d < 8; ++d)
        f[i * 8 + d] = x * w_in[d * 3] + y * w_in[d * 3 + 1] + z * w_in[d * 3 + 2]
                     + b_in[d];
}

// ---------------------------------------------------------------------------
// K2: exact KNN, one wave per query row. Distances in LDS, 32x min-extraction
// with wave-wide argmin (tie -> smaller index, matching lax.top_k order).
// ---------------------------------------------------------------------------
__global__ __launch_bounds__(64) void knn_kernel(
    const float* __restrict__ opc, const float* __restrict__ sqn,
    int* __restrict__ idx)
{
    __shared__ float dl[2][N_];                 // 32 KB
    const int w = threadIdx.x >> 5, lane = threadIdx.x & 31;
    const int row = blockIdx.x * 2 + w;         // b*N + n
    const int b = row / N_, n = row % N_;
    const float* base = opc + (size_t)b * N_ * 3;
    const float px = base[n * 3], py = base[n * 3 + 1], pz = base[n * 3 + 2];
    const float sn = sqn[row];
    for (int j = 0; j < N_ / 32; ++j) {
        int m = j * 32 + lane;
        float qx = base[m * 3], qy = base[m * 3 + 1], qz = base[m * 3 + 2];
        dl[w][m] = sn + sqn[b * N_ + m] - 2.f * (px * qx + py * qy + pz * qz);
    }
    __syncthreads();
    for (int k = 0; k < K_; ++k) {
        float bv = 3.4e38f; int bi = 0x7fffffff;
        for (int j = 0; j < N_ / 32; ++j) {
            int m = j * 32 + lane;
            float v = dl[w][m];
            if (v < bv || (v == bv && m < bi)) { bv = v; bi = m; }
        }
#pragma unroll
        for (int off = 16; off >= 1; off >>= 1) {
            float ov = __shfl_xor(bv, off, 32);
            int   oi = __shfl_xor(bi, off, 32);
            if (ov < bv || (ov == bv && oi < bi)) { bv = ov; bi = oi; }
        }
        if (lane == 0) { idx[(size_t)row * K_ + k] = bi; dl[w][bi] = 3.4e38f; }
        __syncthreads();
    }
}

// ---------------------------------------------------------------------------
// K3: graph feature + 16->64 linear, store f16, accumulate GN sums/sumsq
// ---------------------------------------------------------------------------
__global__ __launch_bounds__(256) void graph_kernel(
    const float* __restrict__ f, const int* __restrict__ idx,
    const float* __restrict__ w_graph, _Float16* __restrict__ g16,
    float* __restrict__ stats)
{
    __shared__ float wg[1024];
    __shared__ float ssum[4], ssq[4];
    const int tid = threadIdx.x;
    for (int i = tid; i < 1024; i += 256) wg[i] = w_graph[i];
    if (tid < 4) { ssum[tid] = 0.f; ssq[tid] = 0.f; }
    __syncthreads();

    const size_t gid = (size_t)blockIdx.x * 256 + tid;  // point id < B*N*K
    const int b  = (int)(gid / ((size_t)N_ * K_));
    const int n  = (int)((gid / K_) % N_);
    const int nb = idx[gid];
    const float* fq = f + ((size_t)b * N_ + n)  * 8;
    const float* fn = f + ((size_t)b * N_ + nb) * 8;
    float feat[16];
#pragma unroll
    for (int c = 0; c < 8; ++c) { float q = fq[c]; feat[c] = fn[c] - q; feat[8 + c] = q; }

    float s[4] = {0, 0, 0, 0}, q2[4] = {0, 0, 0, 0};
    for (int d8 = 0; d8 < 8; ++d8) {
        h8 pack;
#pragma unroll
        for (int u = 0; u < 8; ++u) {
            int d = d8 * 8 + u;
            float acc = 0.f;
#pragma unroll
            for (int c = 0; c < 16; ++c) acc += feat[c] * wg[d * 16 + c];
            pack[u] = (_Float16)acc;
            int grp = d >> 4;
            s[grp] += acc; q2[grp] += acc * acc;
        }
        *(h8*)(g16 + gid * C_ + d8 * 8) = pack;
    }
#pragma unroll
    for (int g = 0; g < 4; ++g) { atomicAdd(&ssum[g], s[g]); atomicAdd(&ssq[g], q2[g]); }
    __syncthreads();
    if (tid < 4) {                                 // whole block shares one batch b
        atomicAdd(&stats[b * 8 + tid],     ssum[tid]);
        atomicAdd(&stats[b * 8 + 4 + tid], ssq[tid]);
    }
}

__global__ void gn_finalize_kernel(float* stats) {
    int t = threadIdx.x;
    if (t >= 16) return;
    int b = t / 4, g = t % 4;
    float cnt = (float)((size_t)N_ * K_ * 16);
    float mu  = stats[b * 8 + g] / cnt;
    float var = stats[b * 8 + 4 + g] / cnt - mu * mu;
    stats[b * 8 + g]     = mu;
    stats[b * 8 + 4 + g] = rsqrtf(var + EPS_);
}

// K5: apply GN + affine + LeakyReLU(0.2) in place on g16
__global__ __launch_bounds__(256) void gn_apply_kernel(
    _Float16* __restrict__ g16, const float* __restrict__ stats,
    const float* __restrict__ gn_g, const float* __restrict__ gn_b)
{
    size_t gid = (size_t)blockIdx.x * 256 + threadIdx.x;
    size_t e0  = gid * 8;                         // 8 f16 per thread
    int    c0  = (int)(e0 & 63);
    size_t pt  = e0 >> 6;
    int    b   = (int)(pt / ((size_t)N_ * K_));
    int    grp = c0 >> 4;
    float mu = stats[b * 8 + grp], rs = stats[b * 8 + 4 + grp];
    h8 v = *(const h8*)(g16 + e0);
    h8 o;
#pragma unroll
    for (int u = 0; u < 8; ++u) {
        int c = c0 + u;
        float x = ((float)v[u] - mu) * rs * gn_g[c] + gn_b[c];
        o[u] = (_Float16)(x >= 0.f ? x : 0.2f * x);
    }
    *(h8*)(g16 + e0) = o;
}

// ---------------------------------------------------------------------------
// K6/K7: 3x3 SAME conv (64->64) as 9 shifted WMMA GEMMs.
// Dynamic LDS (117 KB of the 320 KB WGP LDS):
//   in_lds = 10 rows x 34 cols (1-col halo each side) x 64ch f16, staged by
//            one TDM descriptor per valid row (contiguous 4 KB copies into
//            col offset 1); halo cols + out-of-batch rows zero-filled by VALU.
//   w_lds  = full 9x64x64 f16 weights, one TDM copy.
// Halo layout keeps A-fragment loads unconditional -> clean ds_load+wmma
// stream (no cndmask/v_nop hazards), EXEC stays all-ones for WMMA.
// ---------------------------------------------------------------------------
__global__ __launch_bounds__(256) void conv3x3_kernel(
    const _Float16* __restrict__ in, const _Float16* __restrict__ wf16,
    const float* __restrict__ scale, const float* __restrict__ shift,
    const _Float16* __restrict__ residual, _Float16* __restrict__ out,
    int do_relu)
{
    constexpr int TH = 8;                               // row tile
    constexpr int ROWH = 34 * 64;                       // LDS halves per row
    extern __shared__ _Float16 smem[];
    _Float16* in_lds = smem;                            // 10*34*64 halves
    _Float16* w_lds  = smem + 10 * ROWH;                // 9*64*64 halves
    const int tilesPerB = N_ / TH;
    const int b   = blockIdx.x / tilesPerB;
    const int n0  = (blockIdx.x % tilesPerB) * TH;
    const int tid = threadIdx.x;
    const int wave = tid >> 5, lane = tid & 31;

    // zero the column halos (cols 0 and 33 of all 10 rows): 160 h8 chunks
    {
        h8 z;
#pragma unroll
        for (int u = 0; u < 8; ++u) z[u] = (_Float16)0.f;
        for (int i = tid; i < 160; i += 256) {
            int row = i / 16, side = (i / 8) & 1, sub = i & 7;
            int col = side ? 33 : 0;
            *(h8*)(in_lds + row * ROWH + col * 64 + sub * 8) = z;
        }
        // zero whole edge rows that fall outside the batch (data cols 1..32)
        if (n0 == 0) {
            for (int i = tid; i < 256; i += 256)
                *(h8*)(in_lds + 64 + i * 8) = z;                  // row 0
        }
        if (n0 == N_ - TH) {
            for (int i = tid; i < 256; i += 256)
                *(h8*)(in_lds + 9 * ROWH + 64 + i * 8) = z;       // row 9
        }
    }
    // TDM staging (wave 0 issues; barrier publishes). One descriptor per row,
    // landing at column offset 1 inside the halo layout, plus full weights.
    if (wave == 0) {
#pragma unroll
        for (int r = 0; r < 10; ++r) {
            const int gn = n0 - 1 + r;
            if (gn >= 0 && gn < N_) {
                tdm_copy_f16(in + ((size_t)b * N_ + gn) * (size_t)(K_ * C_),
                             in_lds + r * ROWH + 64,
                             (unsigned)(K_ * C_));
            }
        }
        tdm_copy_f16(wf16, w_lds, 9u * 64u * 64u);
        __builtin_amdgcn_s_wait_tensorcnt(0);
    }
    __syncthreads();

    const int m = lane & 15, hi = lane >> 4;

    for (int cc = 0; cc < 4; ++cc) {                    // 16-out-channel chunks
        const int   co = cc * 16 + m;
        const float sc = scale[co], sh = shift[co];

        for (int pt = wave; pt < 16; pt += 8) {         // 16-pixel tiles
            v8f acc;
#pragma unroll
            for (int u = 0; u < 8; ++u) acc[u] = 0.f;
            const int pixA = pt * 16 + m;               // A row = pixel
            const int prA = pixA >> 5, pcA = pixA & 31;
#pragma unroll
            for (int tap = 0; tap < 9; ++tap) {
                const int dy = tap / 3, dx = tap % 3;
                const int abase = ((prA + dy) * 34 + (pcA + dx)) * 64;
                const _Float16* wb = w_lds + (tap * 64 + co) * 64;
#pragma unroll
                for (int kk = 0; kk < 2; ++kk) {
                    v16h A  = load_bfrag(in_lds + abase + kk * 32 + hi * 16);
                    v16h Bf = load_bfrag(wb + kk * 32 + hi * 16);
                    acc = wmma16(A, Bf, acc);
                }
            }
#pragma unroll
            for (int r = 0; r < 8; ++r) {               // D: M = r + hi*8
                int pix = pt * 16 + (hi ? r + 8 : r);
                int pr = pix >> 5, pcc = pix & 31;
                size_t gpix = ((size_t)b * N_ + (n0 + pr)) * K_ + pcc;
                float x = acc[r] * sc + sh;
                if (residual) x += (float)residual[gpix * C_ + co];
                if (do_relu) x = fmaxf(x, 0.f);
                out[gpix * C_ + co] = (_Float16)x;
            }
        }
    }
}

// ---------------------------------------------------------------------------
// K8: t = h2 . w_blk^T + b_blk via WMMA (K=64, fragments straight from global)
// ---------------------------------------------------------------------------
__global__ __launch_bounds__(256) void blk_gemm_kernel(
    const _Float16* __restrict__ in, const _Float16* __restrict__ w16,
    const float* __restrict__ bias, _Float16* __restrict__ out)
{
    const int wave = threadIdx.x >> 5, lane = threadIdx.x & 31;
    const int m = lane & 15, hi = lane >> 4;
    const size_t p0 = ((size_t)blockIdx.x * 8 + wave) * 16;   // pixel tile base
    v16h A[2];
#pragma unroll
    for (int kk = 0; kk < 2; ++kk)
        A[kk] = load_bfrag(in + (p0 + m) * C_ + kk * 32 + hi * 16);
#pragma unroll
    for (int cc = 0; cc < 4; ++cc) {
        v8f acc;
#pragma unroll
        for (int u = 0; u < 8; ++u) acc[u] = 0.f;
#pragma unroll
        for (int kk = 0; kk < 2; ++kk) {
            v16h Bf = load_bfrag(w16 + ((cc * 16 + m) * 64 + kk * 32 + hi * 16));
            acc = wmma16(A[kk], Bf, acc);
        }
        const int co = cc * 16 + m;
        const float bv = bias[co];
#pragma unroll
        for (int r = 0; r < 8; ++r) {
            size_t pix = p0 + (hi ? r + 8 : r);
            out[pix * C_ + co] = (_Float16)(acc[r] + bv);
        }
    }
}

// ---------------------------------------------------------------------------
// K9: head: y = t . w_img^T + b_img, max over K (lane=k), sigmoid, normalize,
// concat with pc -> d_out (B,N,6) f32
// ---------------------------------------------------------------------------
__global__ __launch_bounds__(256) void head_kernel(
    const _Float16* __restrict__ t16, const float* __restrict__ w_img,
    const float* __restrict__ b_img, const float* __restrict__ pc,
    float* __restrict__ out)
{
    __shared__ float wi[192];
    const int tid = threadIdx.x;
    for (int i = tid; i < 192; i += 256) wi[i] = w_img[i];
    __syncthreads();
    const int wave = tid >> 5, lane = tid & 31;
    const size_t rown = (size_t)blockIdx.x * 8 + wave;    // b*N + n
    const _Float16* tp = t16 + (rown * K_ + lane) * C_;
    float y0 = b_img[0], y1 = b_img[1], y2 = b_img[2];
#pragma unroll
    for (int c8 = 0; c8 < 8; ++c8) {
        h8 v = *(const h8*)(tp + c8 * 8);
#pragma unroll
        for (int u = 0; u < 8; ++u) {
            float x = (float)v[u]; int c = c8 * 8 + u;
            y0 += x * wi[c]; y1 += x * wi[64 + c]; y2 += x * wi[128 + c];
        }
    }
#pragma unroll
    for (int off = 16; off >= 1; off >>= 1) {
        y0 = fmaxf(y0, __shfl_xor(y0, off, 32));
        y1 = fmaxf(y1, __shfl_xor(y1, off, 32));
        y2 = fmaxf(y2, __shfl_xor(y2, off, 32));
    }
    if (lane == 0) {
        float c0 = 1.f / (1.f + expf(-y0));
        float c1 = 1.f / (1.f + expf(-y1));
        float c2 = 1.f / (1.f + expf(-y2));
        out[rown * 6 + 0] = pc[rown * 3 + 0];
        out[rown * 6 + 1] = pc[rown * 3 + 1];
        out[rown * 6 + 2] = pc[rown * 3 + 2];
        out[rown * 6 + 3] = (c0 - 0.485f) / 0.229f;
        out[rown * 6 + 4] = (c1 - 0.456f) / 0.224f;
        out[rown * 6 + 5] = (c2 - 0.406f) / 0.225f;
    }
}

// ---------------------------------------------------------------------------
extern "C" void kernel_launch(void* const* d_in, const int* in_sizes, int n_in,
                              void* d_out, int out_size, void* d_ws, size_t ws_size,
                              hipStream_t stream)
{
    (void)in_sizes; (void)n_in; (void)out_size; (void)ws_size;
    const float* opc     = (const float*)d_in[0];
    const float* pc      = (const float*)d_in[1];
    const float* w_in    = (const float*)d_in[2];
    const float* b_in    = (const float*)d_in[3];
    const float* w_graph = (const float*)d_in[4];
    const float* gn_g    = (const float*)d_in[5];
    const float* gn_b    = (const float*)d_in[6];
    const float* w_c1    = (const float*)d_in[7];
    const float* bn1_g   = (const float*)d_in[8];
    const float* bn1_b   = (const float*)d_in[9];
    const float* bn1_m   = (const float*)d_in[10];
    const float* bn1_v   = (const float*)d_in[11];
    const float* w_c2    = (const float*)d_in[12];
    const float* bn2_g   = (const float*)d_in[13];
    const float* bn2_b   = (const float*)d_in[14];
    const float* bn2_m   = (const float*)d_in[15];
    const float* bn2_v   = (const float*)d_in[16];
    const float* w_blk   = (const float*)d_in[17];
    const float* b_blk   = (const float*)d_in[18];
    const float* w_img   = (const float*)d_in[19];
    const float* b_img   = (const float*)d_in[20];
    float* out = (float*)d_out;

    // ---- workspace layout (256B-aligned chunks), ~204 MB total ----
    char* wsp = (char*)d_ws;
    auto take = [&](size_t sz) -> char* {
        char* p = wsp; wsp += (sz + 255) & ~(size_t)255; return p;
    };
    const size_t NPIX = (size_t)B_ * N_ * K_;            // 524288
    float*    f     = (float*)take((size_t)B_ * N_ * 8 * 4);
    float*    sqn   = (float*)take((size_t)B_ * N_ * 4);
    float*    stats = (float*)take(256);
    float*    bnp   = (float*)take(1024);
    int*      idx   = (int*)take(NPIX * 4);
    _Float16* wc1h  = (_Float16*)take(36864 * 2);
    _Float16* wc2h  = (_Float16*)take(36864 * 2);
    _Float16* wblkh = (_Float16*)take(4096 * 2);
    _Float16* g16   = (_Float16*)take(NPIX * C_ * 2);
    _Float16* h1    = (_Float16*)take(NPIX * C_ * 2);    // also reused for t
    _Float16* h2    = (_Float16*)take(NPIX * C_ * 2);
    _Float16* t16   = h1;                                 // alias: h1 dead after conv2

    const size_t conv_lds = (10 * 34 * 64 + 9 * 64 * 64) * sizeof(_Float16); // 117248

    prep_weights_kernel<<<(77824 + 255) / 256, 256, 0, stream>>>(
        w_c1, w_c2, w_blk, wc1h, wc2h, wblkh);
    prep_bn_kernel<<<1, 128, 0, stream>>>(
        bn1_g, bn1_b, bn1_m, bn1_v, bn2_g, bn2_b, bn2_m, bn2_v, bnp, stats);
    fsq_kernel<<<(B_ * N_ + 255) / 256, 256, 0, stream>>>(opc, w_in, b_in, f, sqn);
    knn_kernel<<<B_ * N_ / 2, 64, 0, stream>>>(opc, sqn, idx);
    graph_kernel<<<(int)(NPIX / 256), 256, 0, stream>>>(f, idx, w_graph, g16, stats);
    gn_finalize_kernel<<<1, 16, 0, stream>>>(stats);
    gn_apply_kernel<<<(int)(NPIX * C_ / 8 / 256), 256, 0, stream>>>(
        g16, stats, gn_g, gn_b);
    conv3x3_kernel<<<B_ * (N_ / 8), 256, conv_lds, stream>>>(
        g16, wc1h, bnp, bnp + 64, nullptr, h1, 1);       // conv1+bn1+relu
    conv3x3_kernel<<<B_ * (N_ / 8), 256, conv_lds, stream>>>(
        h1, wc2h, bnp + 128, bnp + 192, g16, h2, 1);     // conv2+bn2+res+relu
    blk_gemm_kernel<<<(int)(NPIX / 16 / 8), 256, 0, stream>>>(h2, wblkh, b_blk, t16);
    head_kernel<<<B_ * N_ / 8, 256, 0, stream>>>(t16, w_img, b_img, pc, out);
}
